// Mamba3Block_47906065219943
// MI455X (gfx1250) — compile-verified, matrix operand order checked by reading, not run on hardware
//
#include <hip/hip_runtime.h>

typedef __attribute__((ext_vector_type(16))) __bf16   v16bf;
typedef __attribute__((ext_vector_type(8)))  __bf16   v8bf;
typedef __attribute__((ext_vector_type(8)))  float    v8f;
typedef __attribute__((ext_vector_type(4)))  unsigned v4u;
typedef __attribute__((ext_vector_type(8)))  unsigned v8u;

// ---------------------------------------------------------------------------
// fp32 -> bf16 flat convert
// ---------------------------------------------------------------------------
__global__ __launch_bounds__(256) void k_convert_bf16(const float* __restrict__ in,
                                                      __bf16* __restrict__ out,
                                                      size_t n) {
  size_t i = (size_t)blockIdx.x * 256 + threadIdx.x;
  if (i < n) out[i] = (__bf16)in[i];
}

// ---------------------------------------------------------------------------
// W [K][N] fp32  ->  Wt [Npad][K] bf16 (rows n >= N zero-filled)
// ---------------------------------------------------------------------------
__global__ __launch_bounds__(256) void k_transpose_bf16(const float* __restrict__ W,
                                                        __bf16* __restrict__ Wt,
                                                        int K, int N) {
  int n = blockIdx.x;   // 0..Npad-1
  for (int k = threadIdx.x; k < K; k += 256) {
    float v = (n < N) ? W[(size_t)k * N + n] : 0.f;
    Wt[(size_t)n * K + k] = (__bf16)v;
  }
}

// ---------------------------------------------------------------------------
// WMMA GEMM with TDM-staged B panel.
// C[M][ldc] (f32) = A[M][K] (bf16) * Bt[Npad][K]^T (bf16)
// Block tile 64x128: 8 waves as 4(M) x 2(N), wave tile 16x64 (4 accumulators).
// Per 128-wide K chunk, one tensor_load_to_lds DMAs the 128x128 bf16 B tile
// (32 KB) into LDS; fragments are read back with ds_load_b128.
// ---------------------------------------------------------------------------
__global__ __launch_bounds__(256) void k_wmma_gemm_bf16_tdm(
    const __bf16* __restrict__ A,
    const __bf16* __restrict__ Bt,
    float* __restrict__ C,
    int Nact, int K, int ldc, int Npad) {
  __shared__ __bf16 btile[128 * 128];   // 32 KB staged B chunk

  const int tid  = threadIdx.x;
  const int wave = tid >> 5;
  const int lane = tid & 31;
  const int wm = wave >> 1;             // 0..3 (M)
  const int wn = wave & 1;              // 0..1 (N)
  const int m0 = blockIdx.x * 64 + wm * 16;
  const int nb = blockIdx.y * 128;
  const int ml   = lane & 15;           // row (A) / col (B,C,D) of this lane
  const int half = lane >> 4;           // K-half selector (16-bit WMMA layout)

  const __bf16* arow = A + (size_t)(m0 + ml) * K;

  v8f acc[4] = {v8f{}, v8f{}, v8f{}, v8f{}};

  for (int kk = 0; kk < K; kk += 128) {
    if (tid < 32) {   // wave 0 issues the DMA (TDM ignores EXEC)
      const unsigned lds_off = (unsigned)(uintptr_t)(&btile[0]);
      const unsigned long long ga =
          (unsigned long long)(uintptr_t)(Bt + (size_t)nb * K + kk);
      // D# group 0: count=1 | lds_addr | global_addr | type=2
      v4u g0;
      g0[0] = 1u;
      g0[1] = lds_off;
      g0[2] = (unsigned)ga;
      g0[3] = ((unsigned)(ga >> 32) & 0x01FFFFFFu) | (2u << 30);
      // D# group 1: data_size=2B; tensor_dim0=K, tensor_dim1=Npad;
      //             tile 128(dim0) x 128(dim1); dim0_stride=K
      v8u g1;
      g1[0] = (1u << 16);                                        // data_size
      g1[1] = ((unsigned)K & 0xFFFFu) << 16;                     // td0[15:0]
      g1[2] = (((unsigned)K >> 16) & 0xFFFFu) |
              (((unsigned)Npad & 0xFFFFu) << 16);                // td0 hi | td1 lo
      g1[3] = (((unsigned)Npad >> 16) & 0xFFFFu) | (128u << 16); // td1 hi | tile_dim0
      g1[4] = 128u;                                              // tile_dim1
      g1[5] = (unsigned)K;                                       // dim0_stride lo
      g1[6] = 0u;
      g1[7] = 0u;
      asm volatile("tensor_load_to_lds %0, %1" :: "s"(g0), "s"(g1) : "memory");
      __builtin_amdgcn_s_wait_tensorcnt(0);
    }
    __syncthreads();

#pragma unroll
    for (int ks = 0; ks < 4; ++ks) {
      const int k32 = kk + ks * 32;
      __builtin_prefetch(arow + k32 + 128, 0, 0);
      v8bf a0 = *(const v8bf*)(arow + k32 + half * 8);
      v8bf a1 = *(const v8bf*)(arow + k32 + 16 + half * 8);
      v16bf av;
#pragma unroll
      for (int i = 0; i < 8; ++i) { av[i] = a0[i]; av[i + 8] = a1[i]; }
#pragma unroll
      for (int sub = 0; sub < 4; ++sub) {
        const __bf16* brow =
            btile + (size_t)(wn * 64 + sub * 16 + ml) * 128 + ks * 32;
        v8bf b0 = *(const v8bf*)(brow + half * 8);
        v8bf b1 = *(const v8bf*)(brow + 16 + half * 8);
        v16bf bv;
#pragma unroll
        for (int i = 0; i < 8; ++i) { bv[i] = b0[i]; bv[i + 8] = b1[i]; }
        acc[sub] = __builtin_amdgcn_wmma_f32_16x16x32_bf16(
            false, av, false, bv, (short)0, acc[sub], false, false);
      }
    }
    __syncthreads();
  }

#pragma unroll
  for (int sub = 0; sub < 4; ++sub) {
    const int col = nb + wn * 64 + sub * 16 + ml;
    if (col < Nact) {
#pragma unroll
      for (int v = 0; v < 8; ++v)
        C[(size_t)(m0 + half * 8 + v) * ldc + col] = acc[sub][v];
    }
  }
}

// ---------------------------------------------------------------------------
// Pure-global WMMA GEMM (known-good): block 32x64, wave tile 16x16.
// ---------------------------------------------------------------------------
__global__ __launch_bounds__(256) void k_wmma_gemm_bf16(
    const __bf16* __restrict__ A,
    const __bf16* __restrict__ Bt,
    float* __restrict__ C,
    int Nact, int K, int ldc) {
  const int tid  = threadIdx.x;
  const int wave = tid >> 5;
  const int lane = tid & 31;
  const int wm = wave >> 2, wn = wave & 3;
  const int m0 = blockIdx.x * 32 + wm * 16;
  const int n0 = blockIdx.y * 64 + wn * 16;
  const int ml   = lane & 15;
  const int half = lane >> 4;

  const __bf16* arow = A  + (size_t)(m0 + ml) * K;
  const __bf16* brow = Bt + (size_t)(n0 + ml) * K;

  v8f acc = {};
  for (int kk = 0; kk < K; kk += 32) {
    __builtin_prefetch(arow + kk + 64, 0, 0);
    __builtin_prefetch(brow + kk + 64, 0, 0);
    v8bf a0 = *(const v8bf*)(arow + kk + half * 8);
    v8bf a1 = *(const v8bf*)(arow + kk + 16 + half * 8);
    v8bf b0 = *(const v8bf*)(brow + kk + half * 8);
    v8bf b1 = *(const v8bf*)(brow + kk + 16 + half * 8);
    v16bf av, bv;
#pragma unroll
    for (int i = 0; i < 8; ++i) {
      av[i] = a0[i]; av[i + 8] = a1[i];
      bv[i] = b0[i]; bv[i + 8] = b1[i];
    }
    acc = __builtin_amdgcn_wmma_f32_16x16x32_bf16(
        false, av, false, bv, (short)0, acc, false, false);
  }

  const int col = n0 + ml;
  if (col < Nact) {
#pragma unroll
    for (int v = 0; v < 8; ++v)
      C[(size_t)(m0 + half * 8 + v) * ldc + col] = acc[v];
  }
}

// ---------------------------------------------------------------------------
// Depthwise causal conv (K=4) + SiLU over 1536 channels; softplus(dt)+dA
// zxbcdt rows are 2836 wide: [z 0..1280) [xBC 1280..2816) [dt 2816..2836)
// ---------------------------------------------------------------------------
__global__ __launch_bounds__(256) void k_conv_silu_dt(
    const float* __restrict__ zxbcdt,
    const float* __restrict__ conv_w,   // [1536][4]
    const float* __restrict__ conv_b,   // [1536]
    const float* __restrict__ dt_bias,  // [20]
    const float* __restrict__ A_log,    // [20]
    float* __restrict__ xBC,            // [B*L][1536]
    float* __restrict__ dtb,            // [B*L][20]
    float* __restrict__ dAb,            // [B*L][20]
    int L) {
  const int row = blockIdx.x;      // b*L + l
  const int l   = row % L;
  for (int c = threadIdx.x; c < 1536; c += 256) {
    float acc = conv_b[c];
#pragma unroll
    for (int k = 0; k < 4; ++k) {
      const int lk = l + k - 3;
      if (lk >= 0)
        acc += zxbcdt[(size_t)(row + k - 3) * 2836 + 1280 + c] * conv_w[c * 4 + k];
    }
    xBC[(size_t)row * 1536 + c] = acc * (1.f / (1.f + expf(-acc)));   // SiLU
  }
  if (threadIdx.x < 20) {
    const int h = threadIdx.x;
    float v  = zxbcdt[(size_t)row * 2836 + 2816 + h] + dt_bias[h];
    float dt = (v > 20.f) ? v : log1pf(expf(v));                       // softplus
    dtb[(size_t)row * 20 + h] = dt;
    dAb[(size_t)row * 20 + h] = expf(dt * (-expf(A_log[h])));          // exp(dt*A)
  }
}

// ---------------------------------------------------------------------------
// Sequential SSM scan. One block per (b,h): state h[64][128] lives in
// registers, 32 floats/thread. x/B/C staged in LDS per timestep.
// ---------------------------------------------------------------------------
__global__ __launch_bounds__(256) void k_ssm_scan(
    const float* __restrict__ xBC,   // [B*L][1536]
    const float* __restrict__ dtb,   // [B*L][20]
    const float* __restrict__ dAb,   // [B*L][20]
    const float* __restrict__ Dp,    // [20]
    float* __restrict__ y,           // [B*L][1280]
    int L) {
  const int bh = blockIdx.x;
  const int b = bh / 20, h = bh % 20;
  const int tid = threadIdx.x;
  const int p  = tid >> 2;      // 0..63
  const int ng = tid & 3;       // n-chunk: [ng*32, ng*32+32)

  __shared__ float xs[64];
  __shared__ float Bs[128];
  __shared__ float Cs[128];

  float hs[32];
#pragma unroll
  for (int j = 0; j < 32; ++j) hs[j] = 0.f;
  const float Dh = Dp[h];

  for (int l = 0; l < L; ++l) {
    const size_t row = (size_t)b * L + l;
    const float* xrow = xBC + row * 1536;
    for (int i = tid; i < 320; i += 256) {
      if (i < 64)       xs[i]       = xrow[h * 64 + i];
      else if (i < 192) Bs[i - 64]  = xrow[1280 + (i - 64)];
      else              Cs[i - 192] = xrow[1408 + (i - 192)];
    }
    __syncthreads();

    const float dA_t = dAb[row * 20 + h];
    const float dt_t = dtb[row * 20 + h];
    const float xv   = xs[p];
    const float dtx  = dt_t * xv;
    const float* Bn = Bs + ng * 32;
    const float* Cn = Cs + ng * 32;
    float yp = 0.f;
#pragma unroll
    for (int j = 0; j < 32; ++j) {
      hs[j] = hs[j] * dA_t + dtx * Bn[j];
      yp += hs[j] * Cn[j];
    }
    yp += __shfl_xor(yp, 1);
    yp += __shfl_xor(yp, 2);
    if (ng == 0) y[row * 1280 + h * 64 + p] = yp + Dh * xv;
    __syncthreads();
  }
}

// ---------------------------------------------------------------------------
// y = y * silu(z); RMSNorm over 1280; * norm_w; -> bf16 for out-proj GEMM
// ---------------------------------------------------------------------------
__global__ __launch_bounds__(256) void k_gate_rmsnorm(
    const float* __restrict__ zxbcdt,   // z in cols [0,1280)
    const float* __restrict__ y,        // [B*L][1280]
    const float* __restrict__ norm_w,   // [1280]
    __bf16* __restrict__ out) {         // [B*L][1280] bf16
  const int row = blockIdx.x;
  const float* zr = zxbcdt + (size_t)row * 2836;
  const float* yr = y + (size_t)row * 1280;

  float vals[5];
  float ss = 0.f;
#pragma unroll
  for (int i = 0; i < 5; ++i) {
    const int c = threadIdx.x + i * 256;
    const float z = zr[c];
    const float g = yr[c] * (z * (1.f / (1.f + expf(-z))));
    vals[i] = g;
    ss += g * g;
  }
#pragma unroll
  for (int off = 16; off > 0; off >>= 1) ss += __shfl_xor(ss, off);

  __shared__ float red[8];
  __shared__ float rstd_sh;
  if ((threadIdx.x & 31) == 0) red[threadIdx.x >> 5] = ss;
  __syncthreads();
  if (threadIdx.x == 0) {
    float tot = 0.f;
#pragma unroll
    for (int i = 0; i < 8; ++i) tot += red[i];
    rstd_sh = rsqrtf(tot / 1280.f + 1e-5f);
  }
  __syncthreads();
  const float rstd = rstd_sh;
#pragma unroll
  for (int i = 0; i < 5; ++i) {
    const int c = threadIdx.x + i * 256;
    out[(size_t)row * 1280 + c] = (__bf16)(vals[i] * rstd * norm_w[c]);
  }
}

// ---------------------------------------------------------------------------
extern "C" void kernel_launch(void* const* d_in, const int* in_sizes, int n_in,
                              void* d_out, int out_size, void* d_ws, size_t ws_size,
                              hipStream_t stream) {
  const float* x       = (const float*)d_in[0];
  const float* W_in    = (const float*)d_in[1];
  const float* conv_w  = (const float*)d_in[2];
  const float* conv_b  = (const float*)d_in[3];
  const float* dt_bias = (const float*)d_in[4];
  const float* A_log   = (const float*)d_in[5];
  const float* Dp      = (const float*)d_in[6];
  const float* norm_w  = (const float*)d_in[7];
  const float* W_out   = (const float*)d_in[8];
  float* out = (float*)d_out;

  constexpr int Bm = 4, L = 4096, DM = 1280;
  constexpr int NPROJ = 2836, NPAD = 2944;      // 2944 = 23 * 128
  constexpr int M = Bm * L;                     // 16384

  char* ws = (char*)d_ws;
  size_t off = 0;
  auto carve = [&](size_t bytes) -> char* {
    char* p = ws + off;
    off += (bytes + 255) & ~(size_t)255;
    return p;
  };
  __bf16* xb     = (__bf16*)carve((size_t)M * DM * 2);       // x bf16; reused as y_bf16
  __bf16* WinT   = (__bf16*)carve((size_t)NPAD * DM * 2);    // W_in^T bf16, padded
  __bf16* WoutT  = (__bf16*)carve((size_t)DM * DM * 2);      // W_out^T bf16
  float*  zxbcdt = (float*)carve((size_t)M * NPROJ * 4);
  float*  xBC    = (float*)carve((size_t)M * 1536 * 4);
  float*  dtb    = (float*)carve((size_t)M * 20 * 4);
  float*  dAb    = (float*)carve((size_t)M * 20 * 4);
  float*  ybuf   = (float*)carve((size_t)M * DM * 4);
  (void)ws_size; (void)in_sizes; (void)n_in; (void)out_size;

  // 1. operand conversion
  {
    size_t n = (size_t)M * DM;
    k_convert_bf16<<<dim3((unsigned)((n + 255) / 256)), dim3(256), 0, stream>>>(x, xb, n);
    k_transpose_bf16<<<dim3(NPAD), dim3(256), 0, stream>>>(W_in, WinT, DM, NPROJ);
    k_transpose_bf16<<<dim3(DM), dim3(256), 0, stream>>>(W_out, WoutT, DM, DM);
  }
  // 2. in_proj GEMM (WMMA bf16, TDM-staged B panel): zxbcdt = x @ W_in
  k_wmma_gemm_bf16_tdm<<<dim3(M / 64, NPAD / 128), dim3(256), 0, stream>>>(
      xb, WinT, zxbcdt, NPROJ, DM, NPROJ, NPAD);
  // 3. conv + SiLU + softplus(dt) + dA
  k_conv_silu_dt<<<dim3(M), dim3(256), 0, stream>>>(
      zxbcdt, conv_w, conv_b, dt_bias, A_log, xBC, dtb, dAb, L);
  // 4. SSM scan (80 blocks = B*H)
  k_ssm_scan<<<dim3(Bm * 20), dim3(256), 0, stream>>>(xBC, dtb, dAb, Dp, ybuf, L);
  // 5. gate + RMSNorm -> bf16 (reuse xb: in_proj GEMM already consumed it)
  __bf16* ybf = xb;
  k_gate_rmsnorm<<<dim3(M), dim3(256), 0, stream>>>(zxbcdt, ybuf, norm_w, ybf);
  // 6. out_proj GEMM (WMMA bf16): out = y @ W_out
  k_wmma_gemm_bf16<<<dim3(M / 32, DM / 64), dim3(256), 0, stream>>>(
      ybf, WoutT, out, DM, DM, DM);
}
